// EideticPhysicsAttention_43860206027248
// MI455X (gfx1250) — matrix operand order repaired and checked by reading, not compiled, not run on hardware
//
#include <hip/hip_runtime.h>
#include <hip/hip_bf16.h>
#include <stdint.h>

// Problem constants (match reference)
#define N_TOK   65536
#define H_HEADS 8
#define D_DIM   64
#define G_SLICES 64
#define HID     512
#define EPS_F   1e-5f
#define ATT_SCALE 0.125f   // 1/sqrt(64)

typedef __attribute__((ext_vector_type(4)))  __bf16 v4bf;
typedef __attribute__((ext_vector_type(8)))  __bf16 v8bf;
typedef __attribute__((ext_vector_type(16))) __bf16 v16bf;
typedef __attribute__((ext_vector_type(4)))  float  v4f;
typedef __attribute__((ext_vector_type(8)))  float  v8f;
typedef __attribute__((ext_vector_type(4)))  unsigned int u32x4;
typedef __attribute__((ext_vector_type(8)))  int    i32x8;
typedef __attribute__((ext_vector_type(4)))  int    i32x4;

#if defined(__HIP_DEVICE_COMPILE__) && __has_builtin(__builtin_amdgcn_tensor_load_to_lds)
#define HAVE_TDM 1
#else
#define HAVE_TDM 0
#endif

// ---------------------------------------------------------------------------
// Tensor Data Mover: 2D tile Global->LDS with optional LDS row padding.
// D# bit layout per CDNA5 ISA.
// Dims/strides in data_size units (bf16: data_size code 1 = 2 bytes).
// pad_interval code: pad after 2^(code+1) DWORDs; pad_amount code: (code+1) DWORDs.
// ---------------------------------------------------------------------------
__device__ __forceinline__ uint32_t lds_addr_of(const void* p) {
  return (uint32_t)(size_t)p;  // flat LDS aperture: low 32 bits = LDS byte addr
}

__device__ __forceinline__ void tdm_issue_2d(
    const void* gsrc, uint32_t lds_byte,
    uint32_t tile_d0, uint32_t tile_d1,
    uint32_t tensor_d0, uint32_t tensor_d1, uint64_t stride_d0,
    uint32_t pad_interval_code, uint32_t pad_amount_code) {
#if HAVE_TDM
  const uint64_t ga = (uint64_t)(size_t)gsrc;
  u32x4 g0;
  g0[0] = 1u;                                            // count=1, user D#
  g0[1] = lds_byte;                                      // lds_addr
  g0[2] = (uint32_t)(ga & 0xffffffffu);                  // global_addr[31:0]
  g0[3] = (uint32_t)((ga >> 32) & 0x1ffffffu) | (2u << 30);  // [56:32] | type=2
  i32x8 g1;
  uint32_t d0 = (1u << 16)                               // data_size = 2B
              | (1u << 20)                               // pad_enable
              | (pad_interval_code << 22)
              | (pad_amount_code << 25);
  g1[0] = (int)d0;
  g1[1] = (int)((tensor_d0 & 0xffffu) << 16);            // tensor_dim0[15:0]
  g1[2] = (int)(((tensor_d0 >> 16) & 0xffffu) | ((tensor_d1 & 0xffffu) << 16));
  g1[3] = (int)(((tensor_d1 >> 16) & 0xffffu) | ((tile_d0 & 0xffffu) << 16));
  g1[4] = (int)(tile_d1 & 0xffffu);                      // tile_dim1; tile_dim2=0
  g1[5] = (int)(uint32_t)(stride_d0 & 0xffffffffu);      // dim0_stride[31:0]
  g1[6] = (int)(uint32_t)((stride_d0 >> 32) & 0xffffu);  // dim0_stride[47:32]
  g1[7] = 0;
  const i32x4 z4 = {0, 0, 0, 0};
  const i32x8 z8 = {0, 0, 0, 0, 0, 0, 0, 0};
  __builtin_amdgcn_tensor_load_to_lds(g0, g1, z4, z4, z8, 0);
#else
  (void)gsrc; (void)lds_byte; (void)tile_d0; (void)tile_d1;
  (void)tensor_d0; (void)tensor_d1; (void)stride_d0;
  (void)pad_interval_code; (void)pad_amount_code;
#endif
}

template <int N>
__device__ __forceinline__ void tdm_wait() {
#if HAVE_TDM
  __builtin_amdgcn_s_wait_tensorcnt(N);
#endif
}

// ---------------------------------------------------------------------------
// WMMA helpers (CDNA5 wave32, 16x16x32 bf16 -> f32)
// ---------------------------------------------------------------------------
__device__ __forceinline__ v8f wmma_bf16(v16bf a, v16bf b, v8f c) {
  return __builtin_amdgcn_wmma_f32_16x16x32_bf16(false, a, false, b, (short)0, c,
                                                 false, false);
}

__device__ __forceinline__ v16bf frag_cat(v8bf lo, v8bf hi) {
  return __builtin_shufflevector(lo, hi, 0, 1, 2, 3, 4, 5, 6, 7,
                                 8, 9, 10, 11, 12, 13, 14, 15);
}

// A fragment (16x32 bf16, row-major source, K contiguous).
__device__ __forceinline__ v16bf load_frag_a(const __bf16* p, int strideM) {
  const int lane = (int)(threadIdx.x & 31u);
  const __bf16* q = p + (lane & 15) * strideM + ((lane >> 4) << 4);
  return frag_cat(*(const v8bf*)q, *(const v8bf*)(q + 8));
}

// B fragment (32x16 bf16) from K-contiguous storage.
__device__ __forceinline__ v16bf load_frag_b_k(const __bf16* p, int strideN) {
  const int lane = (int)(threadIdx.x & 31u);
  const __bf16* q = p + (lane & 15) * strideN + ((lane >> 4) << 4);
  return frag_cat(*(const v8bf*)q, *(const v8bf*)(q + 8));
}

// B fragment from fragment-ready global storage: [lane][16] contiguous.
__device__ __forceinline__ v16bf load_frag_b_g(const __bf16* p) {
  const int lane = (int)(threadIdx.x & 31u);
  const __bf16* q = p + (lane << 4);
  return frag_cat(*(const v8bf*)q, *(const v8bf*)(q + 8));
}

// 4-element load (+convert) for LDS A-tile fills
__device__ __forceinline__ v4bf load4_cvt(const float* p) {
  const v4f t = *(const v4f*)p;
  v4bf r;
  r[0] = (__bf16)t[0]; r[1] = (__bf16)t[1];
  r[2] = (__bf16)t[2]; r[3] = (__bf16)t[3];
  return r;
}
__device__ __forceinline__ v4bf load4_cvt(const __bf16* p) {
  return *(const v4bf*)p;
}

// ---------------------------------------------------------------------------
// Math helpers
// ---------------------------------------------------------------------------
__device__ __forceinline__ float gelu_tanh(float x) {
  const float x3 = x * x * x;
  return 0.5f * x * (1.0f + tanhf(0.7978845608028654f * (x + 0.044715f * x3)));
}

// Deterministic per-element uniform in [0,1). (JAX threefry not reproduced.)
__device__ __forceinline__ float urand01(uint32_t i) {
  uint32_t s = i * 747796405u + 2891336453u;
  s = ((s >> ((s >> 28) + 4u)) ^ s) * 277803737u;
  s = (s >> 22) ^ s;
  return (float)(s >> 8) * (1.0f / 16777216.0f);
}

// Fragment-ready B index: B[k][n] stored at ((kb*NB + nb)*32 + lane)*16 + j
__device__ __forceinline__ int bfrag_index(int k, int n, int NB) {
  const int kb = k >> 5;
  const int nb = n >> 4;
  const int lane = (n & 15) | (((k >> 4) & 1) << 4);
  const int j = k & 15;
  return (((kb * NB + nb) << 5) + lane) * 16 + j;
}

// ---------------------------------------------------------------------------
// K0: weight prep into fragment-ready bf16 layouts + accumulator zeroing
// ---------------------------------------------------------------------------
__global__ __launch_bounds__(256) void k_prep(
    const float* __restrict__ Wx, const float* __restrict__ Wout,
    const float* __restrict__ Wslice, const float* __restrict__ Wt1,
    __bf16* __restrict__ WxF, __bf16* __restrict__ WoutF,
    __bf16* __restrict__ WsF, __bf16* __restrict__ Wt1F,
    float* __restrict__ norm_acc, float* __restrict__ tok_acc) {
  const int idx = (int)(blockIdx.x * 256 + threadIdx.x);
  for (int i = idx; i < HID * HID; i += (int)(gridDim.x * 256)) {
    const int n = i >> 9, k = i & 511;
    const int d = bfrag_index(k, n, HID / 16);
    WxF[d]   = (__bf16)Wx[i];
    WoutF[d] = (__bf16)Wout[i];
  }
  if (idx < G_SLICES * D_DIM) {
    const int g = idx >> 6, dd = idx & 63;
    const int d = bfrag_index(dd, g, G_SLICES / 16);
    WsF[d]  = (__bf16)Wslice[idx];
    Wt1F[d] = (__bf16)Wt1[idx];
  }
  if (idx < H_HEADS * G_SLICES) norm_acc[idx] = 0.0f;
  if (idx < H_HEADS * G_SLICES * D_DIM) tok_acc[idx] = 0.0f;
}

// ---------------------------------------------------------------------------
// K1 / K6: 128x128-tile GEMM, K=512:  out = A @ B (+ bias)
//   A staged to LDS (TDM double-buffered when A is bf16); B fragment-ready.
// ---------------------------------------------------------------------------
#define AS_STRIDE 40  // 32 K + 8 pad (pad = 4 DW -> code 3; row 16 DW -> code 3)
template <typename TA, bool FINAL, bool USE_TDM>
__global__ __launch_bounds__(256) void k_gemm512(
    const TA* __restrict__ A, const __bf16* __restrict__ Bf,
    const float* __restrict__ bias, float* __restrict__ outF,
    __bf16* __restrict__ outB) {
  __shared__ __bf16 As[2][128 * AS_STRIDE];
  const int m0 = (int)blockIdx.x << 7;
  const int nbb = (int)blockIdx.y << 3;
  const int tid = (int)threadIdx.x;
  const int wave = tid >> 5;
  constexpr bool kTdm = USE_TDM && (HAVE_TDM != 0);
  v8f acc[8] = {};
  if (kTdm && tid < 32) {  // prime the pipeline: tile kb=0 -> buf0
    tdm_issue_2d(&A[(size_t)m0 * HID], lds_addr_of(&As[0][0]),
                 32, 128, HID, N_TOK, HID, 3, 3);
  }
  for (int kb = 0; kb < HID / 32; ++kb) {
    const int k0 = kb << 5;
    const int cur = kTdm ? (kb & 1) : 0;
    if (kb + 1 < HID / 32) {  // prefetch next fragment-ready B slab
      __builtin_prefetch(&Bf[(((kb + 1) * (HID / 16) + nbb) << 9) + (tid << 5)], 0, 0);
    }
    if (kTdm) {
      if (tid < 32) {
        if (kb + 1 < HID / 32) {
          tdm_issue_2d(&A[(size_t)m0 * HID + ((kb + 1) << 5)],
                       lds_addr_of(&As[(kb + 1) & 1][0]),
                       32, 128, HID, N_TOK, HID, 3, 3);
          tdm_wait<1>();   // in-order: current tile has landed
        } else {
          tdm_wait<0>();
        }
      }
    } else {
      for (int i = tid; i < 128 * 8; i += 256) {
        const int r = i >> 3, cg = i & 7;
        *(v4bf*)&As[0][r * AS_STRIDE + (cg << 2)] =
            load4_cvt(&A[(size_t)(m0 + r) * HID + k0 + (cg << 2)]);
      }
    }
    __syncthreads();
    const v16bf a = load_frag_a(&As[cur][(wave << 4) * AS_STRIDE], AS_STRIDE);
#pragma unroll
    for (int ns = 0; ns < 8; ++ns) {
      const v16bf b = load_frag_b_g(&Bf[((kb * (HID / 16) + nbb + ns) << 9)]);
      acc[ns] = wmma_bf16(a, b, acc[ns]);
    }
    __syncthreads();
  }
  const int lane = tid & 31;
  const int mb = m0 + (wave << 4) + ((lane >> 4) << 3);
  const int nb = (nbb << 4) + (lane & 15);
#pragma unroll
  for (int ns = 0; ns < 8; ++ns) {
    const int col = nb + (ns << 4);
    const float bv = bias[col];
#pragma unroll
    for (int v = 0; v < 8; ++v) {
      const size_t o = (size_t)(mb + v) * HID + col;
      const float val = acc[ns][v] + bv;
      if (FINAL) outF[o] = val;
      else       outB[o] = (__bf16)val;
    }
  }
}

// ---------------------------------------------------------------------------
// K2: router — temperature MLP + slice logits + gumbel-softmax + norm
// ---------------------------------------------------------------------------
#define XA_STRIDE 72  // 64 + 8 pad (row 32 DW -> code 4; pad 4 DW -> code 3)
__global__ __launch_bounds__(256) void k_router(
    const __bf16* __restrict__ xp, const __bf16* __restrict__ Wt1F,
    const __bf16* __restrict__ WsF, const float* __restrict__ b_t1,
    const float* __restrict__ W_t2, const float* __restrict__ b_t2,
    const float* __restrict__ t_bias, const float* __restrict__ b_slice,
    __bf16* __restrict__ w_out, float* __restrict__ norm_acc) {
  const int h  = (int)blockIdx.y;
  const int n0 = (int)blockIdx.x << 7;
  const int tid = (int)threadIdx.x;
  const int wave = tid >> 5;
  __shared__ __bf16 xa[128 * XA_STRIDE];
  __shared__ float  sbuf[128][65];
  __shared__ float  tclip[128];

#if HAVE_TDM
  if (tid < 32) {
    tdm_issue_2d(&xp[(size_t)n0 * HID + (h << 6)], lds_addr_of(xa),
                 64, 128, HID, N_TOK, HID, 4, 3);
    tdm_wait<0>();
  }
#else
  for (int i = tid; i < 128 * 16; i += 256) {
    const int r = i >> 4, cg = i & 15;
    *(v4bf*)&xa[r * XA_STRIDE + (cg << 2)] =
        load4_cvt(&xp[(size_t)(n0 + r) * HID + (h << 6) + (cg << 2)]);
  }
#endif
  __syncthreads();

  // GEMM1: s1 = gelu(xa @ Wt1 + b_t1)  (M=128, N=64, K=64)
  {
    v8f acc[4] = {};
#pragma unroll
    for (int ks = 0; ks < 2; ++ks) {
      const v16bf a = load_frag_a(&xa[(wave << 4) * XA_STRIDE + (ks << 5)], XA_STRIDE);
#pragma unroll
      for (int ns = 0; ns < 4; ++ns) {
        const v16bf b = load_frag_b_g(&Wt1F[(ks * 4 + ns) << 9]);
        acc[ns] = wmma_bf16(a, b, acc[ns]);
      }
    }
    const int lane = tid & 31;
    const int mb = (wave << 4) + ((lane >> 4) << 3);
    const int nb = lane & 15;
#pragma unroll
    for (int ns = 0; ns < 4; ++ns) {
      const int col = (ns << 4) + nb;
      const float bv = b_t1[col];
#pragma unroll
      for (int v = 0; v < 8; ++v)
        sbuf[mb + v][col] = gelu_tanh(acc[ns][v] + bv);
    }
  }
  __syncthreads();

  if (tid < 128) {  // per-token temperature
    float s = 0.0f;
#pragma unroll 8
    for (int g = 0; g < G_SLICES; ++g) s += sbuf[tid][g] * W_t2[g];
    const float tt = gelu_tanh(s + b_t2[0]) + t_bias[h];
    tclip[tid] = fmaxf(tt, 0.01f);
  }
  __syncthreads();

  // GEMM2: logits = xa @ Wslice + b_slice
  {
    v8f acc[4] = {};
#pragma unroll
    for (int ks = 0; ks < 2; ++ks) {
      const v16bf a = load_frag_a(&xa[(wave << 4) * XA_STRIDE + (ks << 5)], XA_STRIDE);
#pragma unroll
      for (int ns = 0; ns < 4; ++ns) {
        const v16bf b = load_frag_b_g(&WsF[(ks * 4 + ns) << 9]);
        acc[ns] = wmma_bf16(a, b, acc[ns]);
      }
    }
    const int lane = tid & 31;
    const int mb = (wave << 4) + ((lane >> 4) << 3);
    const int nb = lane & 15;
#pragma unroll
    for (int ns = 0; ns < 4; ++ns) {
      const int col = (ns << 4) + nb;
      const float bv = b_slice[col];
#pragma unroll
      for (int v = 0; v < 8; ++v)
        sbuf[mb + v][col] = acc[ns][v] + bv;
    }
  }
  __syncthreads();

  if (tid < 128) {  // gumbel-softmax per token
    const int n = n0 + tid;
    const uint32_t base = ((uint32_t)h * (uint32_t)N_TOK + (uint32_t)n) * (uint32_t)G_SLICES;
    const float tinv = 1.0f / tclip[tid];
    float mx = -1e30f;
    for (int g = 0; g < G_SLICES; ++g) {
      const float u  = urand01(base + (uint32_t)g);
      const float gn = -logf(-logf(u + 1e-8f) + 1e-8f);
      const float l  = (sbuf[tid][g] + gn) * tinv;
      sbuf[tid][g] = l;
      mx = fmaxf(mx, l);
    }
    float ssum = 0.0f;
    for (int g = 0; g < G_SLICES; ++g) {
      const float e = __expf(sbuf[tid][g] - mx);
      sbuf[tid][g] = e;
      ssum += e;
    }
    const float inv = 1.0f / ssum;
    const size_t wbase = ((size_t)h * N_TOK + (size_t)n) * G_SLICES;
    for (int g = 0; g < G_SLICES; ++g) {
      const float wv = sbuf[tid][g] * inv;
      sbuf[tid][g] = wv;
      w_out[wbase + g] = (__bf16)wv;
    }
  }
  __syncthreads();

  if (tid < G_SLICES) {
    float s = 0.0f;
    for (int r = 0; r < 128; ++r) s += sbuf[r][tid];
    atomicAdd(&norm_acc[(h << 6) + tid], s);
  }
}

// ---------------------------------------------------------------------------
// K3: pooling — tokens[h,g,d] += sum_n x_mid[h,n,d]*w[h,n,g]
//   tiles transposed at fill time (TDM cannot transpose) -> K-contiguous LDS
// ---------------------------------------------------------------------------
#define TP_STRIDE 40
__global__ __launch_bounds__(256) void k_pool(
    const __bf16* __restrict__ w, const __bf16* __restrict__ xp,
    float* __restrict__ tok_acc) {
  const int h = (int)blockIdx.y;
  const int nbeg = (int)blockIdx.x << 10;   // 1024 tokens / block
  const int tid = (int)threadIdx.x;
  const int wave = tid >> 5;
  const int msub  = wave >> 1;
  const int nsub0 = (wave & 1) << 1;
  __shared__ __bf16 wlT[64 * TP_STRIDE];    // [g][token]
  __shared__ __bf16 xlT[64 * TP_STRIDE];    // [d][token]
  v8f acc[2] = {};
  for (int kt = 0; kt < 1024; kt += 32) {
    {
      const int r = tid >> 3, cg = tid & 7;
      const int n = nbeg + kt + r;
      const v8bf wv = *(const v8bf*)&w[((size_t)h * N_TOK + n) * G_SLICES + (cg << 3)];
      const v8bf xv = *(const v8bf*)&xp[(size_t)n * HID + (h << 6) + (cg << 3)];
#pragma unroll
      for (int e = 0; e < 8; ++e) {
        wlT[((cg << 3) + e) * TP_STRIDE + r] = wv[e];
        xlT[((cg << 3) + e) * TP_STRIDE + r] = xv[e];
      }
    }
    __syncthreads();
    const v16bf a = load_frag_a(&wlT[(msub << 4) * TP_STRIDE], TP_STRIDE);
#pragma unroll
    for (int t = 0; t < 2; ++t) {
      const v16bf b = load_frag_b_k(&xlT[((nsub0 + t) << 4) * TP_STRIDE], TP_STRIDE);
      acc[t] = wmma_bf16(a, b, acc[t]);
    }
    __syncthreads();
  }
  const int lane = tid & 31;
  const int gb = (msub << 4) + ((lane >> 4) << 3);
  const int db = lane & 15;
#pragma unroll
  for (int t = 0; t < 2; ++t) {
#pragma unroll
    for (int v = 0; v < 8; ++v) {
      const int g = gb + v, d = ((nsub0 + t) << 4) + db;
      atomicAdd(&tok_acc[(((h << 6) + g) << 6) + d], acc[t][v]);
    }
  }
}

// ---------------------------------------------------------------------------
// K4: tiny per-head attention over G=64 slice tokens; o in fragment layout
// ---------------------------------------------------------------------------
__global__ __launch_bounds__(256) void k_attn(
    const float* __restrict__ tok_acc, const float* __restrict__ norm_acc,
    const float* __restrict__ Wq, const float* __restrict__ Wk,
    const float* __restrict__ Wv, const float* __restrict__ Wo,
    __bf16* __restrict__ oF) {
  const int h = (int)blockIdx.x;
  const int tid = (int)threadIdx.x;
  __shared__ __bf16 tok[64][64];
  __shared__ float  qb[64][64];
  __shared__ float  kb[64][64];
  __shared__ float  ab[64][64];
  for (int i = tid; i < 4096; i += 256) {
    const int g = i >> 6;
    tok[0][i] = (__bf16)(tok_acc[(h << 12) + i] / (norm_acc[(h << 6) + g] + EPS_F));
  }
  __syncthreads();
  for (int i = tid; i < 4096; i += 256) {
    const int g = i >> 6, d = i & 63;
    float sq = 0.f, sk = 0.f;
    for (int e = 0; e < 64; ++e) {
      const float te = (float)tok[g][e];
      sq += te * Wq[(d << 6) + e];
      sk += te * Wk[(d << 6) + e];
    }
    qb[g][d] = sq; kb[g][d] = sk;
  }
  __syncthreads();
  for (int i = tid; i < 4096; i += 256) {
    const int g = i >> 6, g2 = i & 63;
    float s = 0.f;
    for (int e = 0; e < 64; ++e) s += qb[g][e] * kb[g2][e];
    ab[g][g2] = s * ATT_SCALE;
  }
  __syncthreads();
  if (tid < 64) {
    float mx = -1e30f;
    for (int g2 = 0; g2 < 64; ++g2) mx = fmaxf(mx, ab[tid][g2]);
    float ss = 0.f;
    for (int g2 = 0; g2 < 64; ++g2) { const float e = __expf(ab[tid][g2] - mx); ab[tid][g2] = e; ss += e; }
    const float inv = 1.0f / ss;
    for (int g2 = 0; g2 < 64; ++g2) ab[tid][g2] *= inv;
  }
  __syncthreads();
  for (int i = tid; i < 4096; i += 256) {
    const int g = i >> 6, d = i & 63;
    float s = 0.f;
    for (int e = 0; e < 64; ++e) s += (float)tok[g][e] * Wv[(d << 6) + e];
    qb[g][d] = s;
  }
  __syncthreads();
  for (int i = tid; i < 4096; i += 256) {
    const int g = i >> 6, d = i & 63;
    float s = 0.f;
    for (int e = 0; e < 64; ++e) s += ab[g][e] * qb[e][d];
    kb[g][d] = s;
  }
  __syncthreads();
  for (int i = tid; i < 4096; i += 256) {
    const int g = i >> 6, d = i & 63;
    float s = 0.f;
    for (int e = 0; e < 64; ++e) s += kb[g][e] * Wo[(d << 6) + e];
    oF[(h << 12) + bfrag_index(g, d, G_SLICES / 16)] = (__bf16)s;
  }
}

// ---------------------------------------------------------------------------
// K5: scatter-back — y[n, h*64+d] = sum_g w[h,n,g]*o[h,g,d]  (WMMA, K=64)
// ---------------------------------------------------------------------------
__global__ __launch_bounds__(256) void k_scatter(
    const __bf16* __restrict__ w, const __bf16* __restrict__ oF,
    __bf16* __restrict__ y_out) {
  const int h  = (int)blockIdx.y;
  const int n0 = (int)blockIdx.x << 7;
  const int tid = (int)threadIdx.x;
  const int wave = tid >> 5;
  __shared__ __bf16 wl[128 * XA_STRIDE];
#if HAVE_TDM
  if (tid < 32) {
    tdm_issue_2d(&w[((size_t)h * N_TOK + n0) * G_SLICES], lds_addr_of(wl),
                 64, 128, 64, N_TOK, 64, 4, 3);
    tdm_wait<0>();
  }
#else
  for (int i = tid; i < 128 * 16; i += 256) {
    const int r = i >> 4, cg = i & 15;
    *(v4bf*)&wl[r * XA_STRIDE + (cg << 2)] =
        load4_cvt(&w[((size_t)h * N_TOK + n0 + r) * G_SLICES + (cg << 2)]);
  }
#endif
  __syncthreads();
  v8f acc[4] = {};
#pragma unroll
  for (int ks = 0; ks < 2; ++ks) {
    const v16bf a = load_frag_a(&wl[(wave << 4) * XA_STRIDE + (ks << 5)], XA_STRIDE);
#pragma unroll
    for (int ns = 0; ns < 4; ++ns) {
      const v16bf b = load_frag_b_g(&oF[(h << 12) + ((ks * 4 + ns) << 9)]);
      acc[ns] = wmma_bf16(a, b, acc[ns]);
    }
  }
  const int lane = tid & 31;
  const int mb = (wave << 4) + ((lane >> 4) << 3);
  const int nb = lane & 15;
#pragma unroll
  for (int ns = 0; ns < 4; ++ns) {
    const int d = (ns << 4) + nb;
#pragma unroll
    for (int v = 0; v < 8; ++v)
      y_out[(size_t)(n0 + mb + v) * HID + (h << 6) + d] = (__bf16)acc[ns][v];
  }
}

// ---------------------------------------------------------------------------
// Host launcher
// ---------------------------------------------------------------------------
extern "C" void kernel_launch(void* const* d_in, const int* in_sizes, int n_in,
                              void* d_out, int out_size, void* d_ws, size_t ws_size,
                              hipStream_t stream) {
  (void)in_sizes; (void)n_in; (void)out_size; (void)ws_size;
  const float* x       = (const float*)d_in[0];
  const float* W_x     = (const float*)d_in[1];
  const float* b_x     = (const float*)d_in[2];
  const float* W_slice = (const float*)d_in[3];
  const float* b_slice = (const float*)d_in[4];
  const float* W_t1    = (const float*)d_in[5];
  const float* b_t1    = (const float*)d_in[6];
  const float* W_t2    = (const float*)d_in[7];
  const float* b_t2    = (const float*)d_in[8];
  const float* t_bias  = (const float*)d_in[9];
  const float* Wq      = (const float*)d_in[10];
  const float* Wk      = (const float*)d_in[11];
  const float* Wv      = (const float*)d_in[12];
  const float* Wo      = (const float*)d_in[13];
  const float* W_out   = (const float*)d_in[14];
  const float* b_out   = (const float*)d_in[15];
  float* out = (float*)d_out;

  uint8_t* ws = (uint8_t*)d_ws;
  size_t off = 0;
  auto alloc = [&](size_t bytes) -> void* {
    void* p = ws + off;
    off = (off + bytes + 255) & ~(size_t)255;
    return p;
  };
  __bf16* WxF     = (__bf16*)alloc((size_t)HID * HID * 2);
  __bf16* WoutF   = (__bf16*)alloc((size_t)HID * HID * 2);
  __bf16* WsF     = (__bf16*)alloc((size_t)G_SLICES * D_DIM * 2);
  __bf16* Wt1F    = (__bf16*)alloc((size_t)G_SLICES * D_DIM * 2);
  float*  normAcc = (float*)alloc((size_t)H_HEADS * G_SLICES * 4);
  float*  tokAcc  = (float*)alloc((size_t)H_HEADS * G_SLICES * D_DIM * 4);
  __bf16* oF      = (__bf16*)alloc((size_t)H_HEADS * G_SLICES * D_DIM * 2);
  __bf16* wbf     = (__bf16*)alloc((size_t)H_HEADS * N_TOK * G_SLICES * 2);
  __bf16* xpbf    = (__bf16*)alloc((size_t)N_TOK * HID * 2);  // later reused as y

  k_prep<<<1024, 256, 0, stream>>>(W_x, W_out, W_slice, W_t1,
                                   WxF, WoutF, WsF, Wt1F, normAcc, tokAcc);
  k_gemm512<float, false, false><<<dim3(N_TOK / 128, HID / 128), 256, 0, stream>>>(
      x, WxF, b_x, nullptr, xpbf);
  k_router<<<dim3(N_TOK / 128, H_HEADS), 256, 0, stream>>>(
      xpbf, Wt1F, WsF, b_t1, W_t2, b_t2, t_bias, b_slice, wbf, normAcc);
  k_pool<<<dim3(64, H_HEADS), 256, 0, stream>>>(wbf, xpbf, tokAcc);
  k_attn<<<H_HEADS, 256, 0, stream>>>(tokAcc, normAcc, Wq, Wk, Wv, Wo, oF);
  k_scatter<<<dim3(N_TOK / 128, H_HEADS), 256, 0, stream>>>(wbf, oF, xpbf);
  k_gemm512<__bf16, true, true><<<dim3(N_TOK / 128, HID / 128), 256, 0, stream>>>(
      xpbf, WoutF, b_out, out, nullptr);
}
